// GroupQueryAttention_67619965108465
// MI455X (gfx1250) — matrix operand order repaired and checked by reading, not compiled, not run on hardware
//
#include <hip/hip_runtime.h>
#include <hip/hip_bf16.h>
#include <math.h>

// ---------------------------------------------------------------------------
// GQA cross-attention + RoPE for MI455X (gfx1250).
// bf16 WMMA pipeline; projection GEMMs use async global->LDS staging
// (ASYNCcnt); flash attention stages K/V tiles with the Tensor Data Mover
// (TENSORcnt). All LDS tiles double-buffered.
// ---------------------------------------------------------------------------

typedef __attribute__((ext_vector_type(16))) __bf16 v16bf;
typedef __attribute__((ext_vector_type(8)))  float  v8f;
typedef __attribute__((ext_vector_type(4)))  unsigned uv4;
typedef __attribute__((ext_vector_type(8)))  unsigned uv8;

#define B_   2
#define TQ_  2048
#define TK_  2048
#define D_   1024
#define H_   16
#define G_   4
#define HD_  64
#define SCALE_ 0.125f                            // 1/sqrt(64), folded into Q
#define ROPE_FREQ_ (9.21034037197618f / 64.0f)   // ln(10000)/HD

union FragAB { v16bf v; uint4 u[2]; };

static __device__ __forceinline__
v8f wmma_bf16(const v16bf& a, const v16bf& b, const v8f& c) {
  // D = A(16x32 bf16) * B(32x16 bf16) + C(16x16 f32)
  return __builtin_amdgcn_wmma_f32_16x16x32_bf16(
      false, a, false, b, (short)0, c, false, false);
}

// 16-byte async copy global -> LDS (GLOBAL_LOAD_ASYNC_TO_LDS_B128, ASYNCcnt).
static __device__ __forceinline__
void async_copy16(const void* gptr, void* lptr) {
  const unsigned lds = (unsigned)(unsigned long long)lptr;
  const unsigned long long ga = (unsigned long long)gptr;
  asm volatile("global_load_async_to_lds_b128 %0, %1, off"
               :: "v"(lds), "v"(ga) : "memory");
}
static __device__ __forceinline__ void wait_async() {
  asm volatile("s_wait_asynccnt 0x0" ::: "memory");
}

// Tensor Data Mover: 2D tile (2-byte elements) global -> LDS, compacted
// row-major at ldsOff.  D# groups per CDNA5 ISA 8.3/8.4 (count=1, type=2,
// data_size=2B, no gather/iterate/pad, workgroup_mask=0).
static __device__ __forceinline__
void tdm_load_2d(const void* gbase, unsigned ldsOff,
                 unsigned tensorDim0, unsigned tensorDim1,
                 unsigned tileDim0, unsigned tileDim1, unsigned stride0)
{
  const unsigned long long ga = (unsigned long long)gbase;
  uv4 g0;
  g0[0] = 1u;                                              // count=1
  g0[1] = ldsOff;                                          // lds_addr
  g0[2] = (unsigned)ga;                                    // global_addr lo
  g0[3] = ((unsigned)(ga >> 32) & 0x01FFFFFFu) | 0x80000000u; // addr hi | type=2
  uv8 g1;
  g1[0] = 0x00010000u;                                     // data_size = 2B
  g1[1] = tensorDim0 << 16;                                // tensor_dim0 lo
  g1[2] = (tensorDim0 >> 16) | (tensorDim1 << 16);         // dim0 hi | dim1 lo
  g1[3] = (tensorDim1 >> 16) | (tileDim0 << 16);           // dim1 hi | tile_dim0
  g1[4] = tileDim1;                                        // tile_dim1 (tile_dim2=0)
  g1[5] = stride0;                                         // tensor_dim0_stride lo
  g1[6] = 0u;                                              // stride0 hi | stride1 lo
  g1[7] = 0u;
  asm volatile("tensor_load_to_lds %0, %1" :: "s"(g0), "s"(g1) : "memory");
}
static __device__ __forceinline__ void wait_tensor() {
  __builtin_amdgcn_s_wait_tensorcnt(0);
}

// ---------------------------------------------------------------------------
// Pre-pass 1: f32 -> bf16 elementwise (activations).
// ---------------------------------------------------------------------------
__global__ __launch_bounds__(256)
void cvt_bf16(const float* __restrict__ x, __bf16* __restrict__ y, int n)
{
  const int i = (blockIdx.x * 256 + threadIdx.x) * 4;
  if (i >= n) return;
  const float4 v = *reinterpret_cast<const float4*>(x + i);
  union { __bf16 h[4]; uint2 u; } p;
  p.h[0] = (__bf16)v.x; p.h[1] = (__bf16)v.y;
  p.h[2] = (__bf16)v.z; p.h[3] = (__bf16)v.w;
  *reinterpret_cast<uint2*>(y + i) = p.u;
}

// ---------------------------------------------------------------------------
// Pre-pass 2: W[K][N] f32 -> Wt[N][K] bf16 (transpose + convert, one-time).
// ---------------------------------------------------------------------------
__global__ __launch_bounds__(256)
void cvt_transpose(const float* __restrict__ W, __bf16* __restrict__ Wt,
                   int K, int N)
{
  const int idx = blockIdx.x * 256 + threadIdx.x;
  if (idx >= K * N) return;
  const int k = idx / N, n = idx % N;          // coalesced read
  Wt[(size_t)n * K + k] = (__bf16)W[idx];
}

// ---------------------------------------------------------------------------
// GEMM: Out = (A[M,K](bf16) * Wt[N,K](bf16)^T + bias) * oscale, opt. RoPE.
// OUTM: 0 -> bf16 [B, N/64, T, 64] | 1 -> bf16 [B, N/64, 64, T] | 2 -> f32 [M,N]
// Block tile 128x64, K-step 32, 8 waves (4x2), 2x2 WMMA per wave.
// Double-buffered LDS, async global->LDS staging.
// ---------------------------------------------------------------------------
template<bool ROPE, int OUTM>
__global__ __launch_bounds__(256)
void proj_gemm(const __bf16* __restrict__ A, const __bf16* __restrict__ Wt,
               const float* __restrict__ bias, void* __restrict__ Out,
               int M, int N, int K, int T, float oscale)
{
  __shared__ __align__(16) __bf16 As[2][128][32];
  __shared__ __align__(16) __bf16 Bs[2][64][32];

  const int tid  = threadIdx.x;
  const int lane = tid & 31;
  const int w    = tid >> 5;
  const int half = lane >> 4;
  const int lm   = lane & 15;

  const int blockM = blockIdx.x * 128;
  const int blockN = blockIdx.y * 64;
  const int waveM  = (w & 3) * 32;
  const int waveN  = (w >> 2) * 32;

  auto stage = [&](int buf, int k0) {
    #pragma unroll
    for (int i = 0; i < 2; ++i) {               // A tile: 512 x 16B
      const int e   = tid + i * 256;
      const int row = e >> 2;
      const int cb  = (e & 3) * 8;
      async_copy16(A + (size_t)(blockM + row) * K + k0 + cb, &As[buf][row][cb]);
    }
    {                                           // B tile: 256 x 16B
      const int n  = tid >> 2;
      const int cb = (tid & 3) * 8;
      async_copy16(Wt + (size_t)(blockN + n) * K + k0 + cb, &Bs[buf][n][cb]);
    }
  };

  v8f acc[2][2] = {};
  const int nk = K / 32;
  stage(0, 0);

  for (int ki = 0; ki < nk; ++ki) {
    wait_async();
    __syncthreads();
    const int cur = ki & 1;
    if (ki + 1 < nk) stage(cur ^ 1, (ki + 1) * 32);

    FragAB af[2], bf[2];
    #pragma unroll
    for (int mt = 0; mt < 2; ++mt) {
      const int m = waveM + mt * 16 + lm;
      af[mt].u[0] = *reinterpret_cast<const uint4*>(&As[cur][m][half * 8]);
      af[mt].u[1] = *reinterpret_cast<const uint4*>(&As[cur][m][16 + half * 8]);
    }
    #pragma unroll
    for (int nt = 0; nt < 2; ++nt) {
      const int n = waveN + nt * 16 + lm;
      bf[nt].u[0] = *reinterpret_cast<const uint4*>(&Bs[cur][n][half * 8]);
      bf[nt].u[1] = *reinterpret_cast<const uint4*>(&Bs[cur][n][16 + half * 8]);
    }
    #pragma unroll
    for (int mt = 0; mt < 2; ++mt)
      #pragma unroll
      for (int nt = 0; nt < 2; ++nt)
        acc[mt][nt] = wmma_bf16(af[mt].v, bf[nt].v, acc[mt][nt]);
  }

  // ---- epilogue: bias (+RoPE), scale, store ----
  #pragma unroll
  for (int mt = 0; mt < 2; ++mt) {
    #pragma unroll
    for (int nt = 0; nt < 2; ++nt) {
      const int nG  = blockN + waveN + nt * 16 + lm;
      const float bia = bias[nG];
      #pragma unroll
      for (int r = 0; r < 8; ++r) {
        const int mG = blockM + waveM + mt * 16 + half * 8 + r;
        float val = acc[mt][nt][r] + bia;
        if constexpr (ROPE) {
          const int   t   = mG % T;
          const int   d   = nG & (HD_ - 1);
          const float ang = (float)t * __expf(-(float)(d & ~1) * ROPE_FREQ_);
          const float cs = __cosf(ang), sn = __sinf(ang);
          const float partner = __shfl_xor(val, 1, 32);
          val = (nG & 1) ? (partner * sn + val * cs)   // oo = xe*sin + xo*cos
                         : (val * cs - partner * sn);  // oe = xe*cos - xo*sin
        }
        val *= oscale;
        const int hh = nG >> 6, dd = nG & 63;
        const int bb = mG / T,  tt = mG % T;
        const int HN = N >> 6;
        if constexpr (OUTM == 0) {
          ((__bf16*)Out)[((size_t)(bb * HN + hh) * T + tt) * HD_ + dd] = (__bf16)val;
        } else if constexpr (OUTM == 1) {
          ((__bf16*)Out)[((size_t)(bb * HN + hh) * HD_ + dd) * T + tt] = (__bf16)val;
        } else {
          ((float*)Out)[(size_t)mG * N + nG] = val;
        }
      }
    }
  }
}

// ---------------------------------------------------------------------------
// Flash attention: grid (TQ/64, B*H), 128 threads (4 waves x 16 queries).
// K chunk [32 keys][64 hd] and V chunk [64 hd][32 keys] staged by the Tensor
// Data Mover (wave 0 issues, TENSORcnt + barrier), double-buffered.
// S via WMMA, online softmax: row-max via shfl_xor butterfly, row-sum via
// WMMA against a ones matrix (broadcast comes for free in the C layout).
// ---------------------------------------------------------------------------
__global__ __launch_bounds__(128)
void flash_attn(const __bf16* __restrict__ Q, const __bf16* __restrict__ Kh,
                const __bf16* __restrict__ Vg, __bf16* __restrict__ O)
{
  __shared__ __align__(16) __bf16 Ks[2][32][64];   // keys x hd
  __shared__ __align__(16) __bf16 Vt[2][64][32];   // hd x keys
  __shared__ __align__(16) __bf16 Ps[4][16][32];   // per-wave P relayout

  const int tid  = threadIdx.x;
  const int lane = tid & 31;
  const int w    = tid >> 5;
  const int half = lane >> 4;
  const int lm   = lane & 15;

  const int bh = blockIdx.y;
  const int b  = bh / H_;
  const int h  = bh % H_;
  const int g  = h / (H_ / G_);
  const int qWave = blockIdx.x * 64 + w * 16;

  const __bf16* Qb = Q  + ((size_t)(b * H_ + h) * TQ_) * HD_;
  const __bf16* Kb = Kh + ((size_t)(b * G_ + g) * TK_) * HD_;
  const __bf16* Vb = Vg + ((size_t)(b * G_ + g) * HD_) * TK_;  // [hd][TK]

  // Q fragments (Q was pre-scaled by 1/sqrt(HD) in the projection).
  FragAB qf[2];
  {
    const __bf16* qrow = Qb + (size_t)(qWave + lm) * HD_;
    #pragma unroll
    for (int kk = 0; kk < 2; ++kk) {
      qf[kk].u[0] = *reinterpret_cast<const uint4*>(qrow + kk * 32 + half * 8);
      qf[kk].u[1] = *reinterpret_cast<const uint4*>(qrow + kk * 32 + 16 + half * 8);
    }
  }

  // ones(32x16) bf16 B-fragment for WMMA row-sum
  FragAB ones;
  ones.u[0] = make_uint4(0x3F803F80u, 0x3F803F80u, 0x3F803F80u, 0x3F803F80u);
  ones.u[1] = ones.u[0];

  auto stage = [&](int buf, int c0) {
    if (w == 0) {
      // K tile: 32 rows x 64 elems, row stride 64 (contiguous rows)
      tdm_load_2d(Kb + (size_t)c0 * HD_, (unsigned)(size_t)&Ks[buf][0][0],
                  HD_, TK_, HD_, 32, HD_);
      // V tile (d-major): 64 rows x 32 elems, row stride TK
      tdm_load_2d(Vb + c0, (unsigned)(size_t)&Vt[buf][0][0],
                  TK_, HD_, 32, HD_, TK_);
    }
  };

  v8f o[4] = {};
  float mrow[8], lrow[8];
  #pragma unroll
  for (int r = 0; r < 8; ++r) { mrow[r] = -INFINITY; lrow[r] = 0.0f; }
  const v8f zero = {};

  const int nc = TK_ / 32;
  stage(0, 0);

  for (int ci = 0; ci < nc; ++ci) {
    if (w == 0) wait_tensor();
    __syncthreads();
    const int cur = ci & 1;
    if (ci + 1 < nc) stage(cur ^ 1, (ci + 1) * 32);

    // ---- S = Q * K^T : hoist all 4 K-fragments, then 4 WMMAs ----
    FragAB kb[2][2];
    #pragma unroll
    for (int nt = 0; nt < 2; ++nt)
      #pragma unroll
      for (int kk = 0; kk < 2; ++kk) {
        kb[nt][kk].u[0] =
            *reinterpret_cast<const uint4*>(&Ks[cur][nt * 16 + lm][kk * 32 + half * 8]);
        kb[nt][kk].u[1] =
            *reinterpret_cast<const uint4*>(&Ks[cur][nt * 16 + lm][kk * 32 + 16 + half * 8]);
      }
    v8f s[2];
    #pragma unroll
    for (int nt = 0; nt < 2; ++nt) {
      v8f acc = zero;
      #pragma unroll
      for (int kk = 0; kk < 2; ++kk)
        acc = wmma_bf16(qf[kk].v, kb[nt][kk].v, acc);
      s[nt] = acc;
    }

    // ---- online softmax: row-max butterfly, exp, store P ----
    float alpha[8];
    #pragma unroll
    for (int r = 0; r < 8; ++r) {
      const float s0 = s[0][r];
      const float s1 = s[1][r];
      float cm = fmaxf(s0, s1);
      #pragma unroll
      for (int msk = 1; msk < 16; msk <<= 1)
        cm = fmaxf(cm, __shfl_xor(cm, msk, 32));
      const float mnew = fmaxf(mrow[r], cm);
      alpha[r] = __expf(mrow[r] - mnew);
      mrow[r]  = mnew;
      Ps[w][half * 8 + r][lm]      = (__bf16)__expf(s0 - mnew);
      Ps[w][half * 8 + r][16 + lm] = (__bf16)__expf(s1 - mnew);
    }

    // rescale running O
    #pragma unroll
    for (int dt = 0; dt < 4; ++dt)
      #pragma unroll
      for (int r = 0; r < 8; ++r)
        o[dt][r] *= alpha[r];

    // P as A-fragment (16x32), wave-private LDS round trip
    FragAB pf;
    pf.u[0] = *reinterpret_cast<const uint4*>(&Ps[w][lm][half * 8]);
    pf.u[1] = *reinterpret_cast<const uint4*>(&Ps[w][lm][16 + half * 8]);

    // row-sum via matrix unit: every lane receives its 8 row sums
    const v8f lsum = wmma_bf16(pf.v, ones.v, zero);
    #pragma unroll
    for (int r = 0; r < 8; ++r)
      lrow[r] = lrow[r] * alpha[r] + lsum[r];

    // ---- O += P * V : hoist all 4 V-fragments, then 4 WMMAs ----
    FragAB vf[4];
    #pragma unroll
    for (int dt = 0; dt < 4; ++dt) {
      vf[dt].u[0] = *reinterpret_cast<const uint4*>(&Vt[cur][dt * 16 + lm][half * 8]);
      vf[dt].u[1] = *reinterpret_cast<const uint4*>(&Vt[cur][dt * 16 + lm][16 + half * 8]);
    }
    #pragma unroll
    for (int dt = 0; dt < 4; ++dt)
      o[dt] = wmma_bf16(pf.v, vf[dt].v, o[dt]);
  }

  // ---- normalize and write bf16 O into [B*TQ, D] for the output GEMM ----
  #pragma unroll
  for (int dt = 0; dt < 4; ++dt) {
    #pragma unroll
    for (int r = 0; r < 8; ++r) {
      const float val = o[dt][r] / lrow[r];
      const int mG  = qWave + half * 8 + r;
      const int col = h * HD_ + dt * 16 + lm;
      O[(size_t)(b * TQ_ + mG) * D_ + col] = (__bf16)val;
    }
  }
}

// ---------------------------------------------------------------------------
extern "C" void kernel_launch(void* const* d_in, const int* in_sizes, int n_in,
                              void* d_out, int out_size, void* d_ws, size_t ws_size,
                              hipStream_t stream)
{
  (void)in_sizes; (void)n_in; (void)out_size; (void)ws_size;

  const float* X   = (const float*)d_in[0];
  const float* Enc = (const float*)d_in[1];
  const float* Wq  = (const float*)d_in[2];
  const float* bq  = (const float*)d_in[3];
  const float* Wk  = (const float*)d_in[4];
  const float* bk  = (const float*)d_in[5];
  const float* Wv  = (const float*)d_in[6];
  const float* bv  = (const float*)d_in[7];
  const float* Wo  = (const float*)d_in[8];
  const float* bo  = (const float*)d_in[9];

  // workspace layout (MB): Xb 8 | Eb 8 | Wqt 2 | Wkt .5 | Wvt .5 | Wot 2 |
  //                        Qh 8 | Kh 2 | Vg 2 | Oh 8    (total 41 MB)
  char* ws = (char*)d_ws;
  const size_t MB = 1024 * 1024;
  __bf16* Xb  = (__bf16*)(ws);
  __bf16* Eb  = (__bf16*)(ws + 8 * MB);
  __bf16* Wqt = (__bf16*)(ws + 16 * MB);
  __bf16* Wkt = (__bf16*)(ws + 18 * MB);
  __bf16* Wvt = (__bf16*)(ws + 18 * MB + 512 * 1024);
  __bf16* Wot = (__bf16*)(ws + 19 * MB);
  __bf16* Qh  = (__bf16*)(ws + 21 * MB);
  __bf16* Kh  = (__bf16*)(ws + 29 * MB);
  __bf16* Vg  = (__bf16*)(ws + 31 * MB);
  __bf16* Oh  = (__bf16*)(ws + 33 * MB);

  const dim3 blk(256);
  const int nX = B_ * TQ_ * D_;     // 4M
  const int nE = B_ * TK_ * D_;     // 4M

  // one-time conversions
  cvt_bf16<<<dim3(nX / 1024), blk, 0, stream>>>(X,   Xb, nX);
  cvt_bf16<<<dim3(nE / 1024), blk, 0, stream>>>(Enc, Eb, nE);
  cvt_transpose<<<dim3((D_ * D_) / 256),        blk, 0, stream>>>(Wq, Wqt, D_, D_);
  cvt_transpose<<<dim3((D_ * G_ * HD_) / 256),  blk, 0, stream>>>(Wk, Wkt, D_, G_ * HD_);
  cvt_transpose<<<dim3((D_ * G_ * HD_) / 256),  blk, 0, stream>>>(Wv, Wvt, D_, G_ * HD_);
  cvt_transpose<<<dim3((D_ * D_) / 256),        blk, 0, stream>>>(Wo, Wot, D_, D_);

  // Q = (X*Wq+bq)*RoPE*scale -> [B,H,TQ,64] bf16  (1/sqrt(HD) folded here)
  proj_gemm<true, 0><<<dim3((B_ * TQ_) / 128, D_ / 64), blk, 0, stream>>>(
      Xb, Wqt, bq, Qh, B_ * TQ_, D_, D_, TQ_, SCALE_);
  // K = enc*Wk+bk, fused RoPE -> [B,G,TK,64] bf16
  proj_gemm<true, 0><<<dim3((B_ * TK_) / 128, (G_ * HD_) / 64), blk, 0, stream>>>(
      Eb, Wkt, bk, Kh, B_ * TK_, G_ * HD_, D_, TK_, 1.0f);
  // V = enc*Wv+bv -> [B,G,64,TK] bf16 (d-major for transpose-free staging)
  proj_gemm<false, 1><<<dim3((B_ * TK_) / 128, (G_ * HD_) / 64), blk, 0, stream>>>(
      Eb, Wvt, bv, Vg, B_ * TK_, G_ * HD_, D_, TK_, 1.0f);
  // flash attention -> [B*TQ, D] bf16
  flash_attn<<<dim3(TQ_ / 64, B_ * H_), dim3(128), 0, stream>>>(Qh, Kh, Vg, Oh);
  // out = O*Wo+bo -> f32
  proj_gemm<false, 2><<<dim3((B_ * TQ_) / 128, D_ / 64), blk, 0, stream>>>(
      Oh, Wot, bo, (float*)d_out, B_ * TQ_, D_, D_, TQ_, 1.0f);
}
